// AttentionBlock_8443905704451
// MI455X (gfx1250) — compile-verified, matrix operand order checked by reading, not run on hardware
//
#include <hip/hip_runtime.h>

// Problem constants (from reference)
constexpr int BATCH = 4;
constexpr int CHN   = 256;
constexpr int SEQ   = 4096;   // H*W
constexpr int HEADS = 4;
constexpr int HDIM  = 64;     // CHN / HEADS
constexpr int GRP   = 32;
constexpr int CPG   = CHN / GRP;  // 8
constexpr float EPS = 1e-5f;
constexpr float QK_SCALE = 0.35355339059327379f;  // 64^-0.25

typedef __attribute__((ext_vector_type(16))) __bf16 v16bf;
typedef __attribute__((ext_vector_type(8)))  float  v8f;

__device__ __forceinline__ __bf16 f2bf(float x) {
  unsigned u = __builtin_bit_cast(unsigned, x);
  unsigned r = u + 0x7FFFu + ((u >> 16) & 1u);   // round-to-nearest-even
  unsigned short h = (unsigned short)(r >> 16);
  return __builtin_bit_cast(__bf16, h);
}

// CDNA5 WMMA 16x16x32 bf16 fragment load.
// ISA 7.12.2 (16-bit 16x32 A): lane = row (M for A, N for B); per-lane 32
// K-values interleaved: lanes 0-15 hold K {0..7,16..23}, lanes 16-31 hold
// K {8..15,24..31}, two packed per VGPR. `row` points at this lane's 32
// contiguous K elements; kHalf = 8 for lanes 16-31 else 0.
// (Compiler vectorizes each fragment into two global_load_b128.)
__device__ __forceinline__ v16bf load_frag(const __bf16* row, int kHalf) {
  v16bf f;
#pragma unroll
  for (int j = 0; j < 8; ++j) {
    int kb = ((j & 4) ? 16 : 0) + kHalf + 2 * (j & 3);
    f[2 * j]     = row[kb];
    f[2 * j + 1] = row[kb + 1];
  }
  return f;
}

__device__ __forceinline__ v16bf load_frag_f32(const float* row, int kHalf) {
  v16bf f;
#pragma unroll
  for (int j = 0; j < 8; ++j) {
    int kb = ((j & 4) ? 16 : 0) + kHalf + 2 * (j & 3);
    f[2 * j]     = f2bf(row[kb]);
    f[2 * j + 1] = f2bf(row[kb + 1]);
  }
  return f;
}

__device__ __forceinline__ v8f wmma_bf16(v16bf a, v16bf b, v8f c) {
#if defined(__HIP_DEVICE_COMPILE__)
  return __builtin_amdgcn_wmma_f32_16x16x32_bf16(false, a, false, b, (short)0, c,
                                                 false, false);
#else
  return c;
#endif
}

// ---------------- Kernel 0: weights fp32 -> bf16 ----------------
__global__ void wconv_kernel(const float* qkv_w, const float* proj_w,
                             __bf16* wqkv, __bf16* wproj) {
  int i = blockIdx.x * 256 + threadIdx.x;
  constexpr int N1 = 3 * CHN * CHN;
  if (i < N1)                 wqkv[i] = f2bf(qkv_w[i]);
  else if (i < N1 + CHN*CHN)  wproj[i - N1] = f2bf(proj_w[i - N1]);
}

// ---------------- Kernel 1: GroupNorm stats (mean, rstd per (b,g)) ----------
__global__ void gn_stats_kernel(const float* x, float* stats) {
  __shared__ float s1[256], s2[256];
  const int bg = blockIdx.x;                       // b*GRP + g
  const size_t base = (size_t)bg * CPG * SEQ;      // channels of a group are contiguous
  float sum = 0.f, sq = 0.f;
  for (int i = threadIdx.x; i < CPG * SEQ; i += 256) {
    float v = x[base + i];
    sum += v; sq += v * v;
  }
  s1[threadIdx.x] = sum; s2[threadIdx.x] = sq;
  __syncthreads();
  for (int off = 128; off; off >>= 1) {
    if (threadIdx.x < (unsigned)off) {
      s1[threadIdx.x] += s1[threadIdx.x + off];
      s2[threadIdx.x] += s2[threadIdx.x + off];
    }
    __syncthreads();
  }
  if (threadIdx.x == 0) {
    const float inv_n = 1.0f / (CPG * SEQ);
    float mean = s1[0] * inv_n;
    float var  = s2[0] * inv_n - mean * mean;
    stats[bg * 2]     = mean;
    stats[bg * 2 + 1] = rsqrtf(var + EPS);
  }
}

// ---------------- Kernel 2: apply GN, write xn transposed [B][T][C] bf16 ----
__global__ void gn_apply_kernel(const float* x, const float* gn_w, const float* gn_b,
                                const float* stats, __bf16* xnT) {
  size_t i = (size_t)blockIdx.x * 256 + threadIdx.x;   // over B*C*T
  int t = (int)(i & (SEQ - 1));
  int c = (int)((i >> 12) & (CHN - 1));
  int b = (int)(i >> 20);
  int bg = b * GRP + (c >> 3);
  float mean = stats[bg * 2], rstd = stats[bg * 2 + 1];
  float v = (x[i] - mean) * rstd * gn_w[c] + gn_b[c];
  xnT[((size_t)b * SEQ + t) * CHN + c] = f2bf(v);
}

// ---------------- Kernel 3: QKV GEMM (M=768, N=4096, K=256 per batch) -------
// 4 waves/block in a 2x2 arrangement; each wave owns a 32x32 register tile
// (2x2 accumulators): per K-step, 2 A-frags + 2 B-frags feed 4 WMMAs.
// Epilogue: +bias, scale q/k, scatter into qT/kT ([BH][T][64]) and v ([BH][64][T]).
__global__ __launch_bounds__(128) void qkv_gemm_kernel(
    const __bf16* wqkv, const float* qkv_b, const __bf16* xnT,
    __bf16* qT, __bf16* kT, __bf16* vB) {
  const int b    = blockIdx.z;
  const int lane = threadIdx.x & 31;
  const int wave = threadIdx.x >> 5;
  const int wm   = wave >> 1, wn = wave & 1;
  const int m0   = blockIdx.y * 64 + wm * 32;
  const int t0   = blockIdx.x * 64 + wn * 32;
  const int n16  = lane & 15;
  const int kHalf = (lane & 16) ? 8 : 0;
  const int mhi   = (lane & 16) ? 8 : 0;

  const __bf16* arow0 = wqkv + (size_t)(m0 + n16) * CHN;            // A rows (W)
  const __bf16* arow1 = arow0 + (size_t)16 * CHN;
  const __bf16* brow0 = xnT + ((size_t)b * SEQ + t0 + n16) * CHN;   // B rows (xn^T)
  const __bf16* brow1 = brow0 + (size_t)16 * CHN;

  v8f acc00 = {}, acc01 = {}, acc10 = {}, acc11 = {};
#pragma unroll
  for (int ks = 0; ks < CHN / 32; ++ks) {
    v16bf a0 = load_frag(arow0 + ks * 32, kHalf);
    v16bf a1 = load_frag(arow1 + ks * 32, kHalf);
    v16bf b0 = load_frag(brow0 + ks * 32, kHalf);
    v16bf b1 = load_frag(brow1 + ks * 32, kHalf);
    acc00 = wmma_bf16(a0, b0, acc00);
    acc01 = wmma_bf16(a0, b1, acc01);
    acc10 = wmma_bf16(a1, b0, acc10);
    acc11 = wmma_bf16(a1, b1, acc11);
  }

  v8f accs[2][2] = {{acc00, acc01}, {acc10, acc11}};
#pragma unroll
  for (int i = 0; i < 2; ++i)
#pragma unroll
    for (int j = 0; j < 2; ++j) {
      const int tcol = t0 + j * 16 + n16;
#pragma unroll
      for (int r = 0; r < 8; ++r) {
        int o = m0 + i * 16 + r + mhi;
        float v = accs[i][j][r] + qkv_b[o];
        if (o < 2 * CHN) v *= QK_SCALE;      // fold attention scale into q and k
        int oo = o & (CHN - 1);
        int h = oo >> 6, cc = oo & 63;
        size_t bh = (size_t)(b * HEADS + h);
        if (o < CHN)            qT[(bh * SEQ + tcol) * HDIM + cc] = f2bf(v);
        else if (o < 2 * CHN)   kT[(bh * SEQ + tcol) * HDIM + cc] = f2bf(v);
        else                    vB[(bh * HDIM + cc) * SEQ + tcol] = f2bf(v);
      }
    }
}

// ---------------- Kernel 4: attention (one block per (bh, 16-query tile)) ---
// LDS: 16x4096 fp32 logits panel (256KB) + 8x16x64 PV partials (32KB) = 288KB,
// exploiting the 320KB WGP LDS of CDNA5. 8 waves. Q fragments reused across
// all 256 key tiles; P fragment reused across 4 V tiles.
__global__ __launch_bounds__(256) void attn_kernel(
    const __bf16* qT, const __bf16* kT, const __bf16* vB, __bf16* aT) {
  extern __shared__ char dynsm[];
  float* Lg   = (float*)dynsm;               // [16][SEQ]
  float* red2 = Lg + 16 * SEQ;               // [8][16][64]
  __shared__ float red[256];

  const int blk  = blockIdx.x;
  const int bh   = blk >> 8;                 // batch*head
  const int tq0  = (blk & 255) * 16;
  const int tid  = threadIdx.x;
  const int lane = tid & 31;
  const int wave = tid >> 5;
  const int n16  = lane & 15;
  const int kHalf = (lane & 16) ? 8 : 0;
  const int mhi   = (lane & 16) ? 8 : 0;

  // Q fragments (A-matrix: M=tq, K=cc in [0,64)), reused for all key tiles.
  const __bf16* qrow = qT + ((size_t)bh * SEQ + tq0 + n16) * HDIM;
  v16bf qa0 = load_frag(qrow, kHalf);
  v16bf qa1 = load_frag(qrow + 32, kHalf);

  // ---- Phase 1: logits = (q*s)·(k*s)^T, 8 waves tile the 4096 keys ----
  for (int pass = 0; pass < SEQ / (16 * 8); ++pass) {
    int s0 = (pass * 8 + wave) * 16;
    const __bf16* krow = kT + ((size_t)bh * SEQ + s0 + n16) * HDIM;  // B: N=s, K=cc
    v8f acc = {};
    acc = wmma_bf16(qa0, load_frag(krow, kHalf), acc);
    acc = wmma_bf16(qa1, load_frag(krow + 32, kHalf), acc);
#pragma unroll
    for (int r = 0; r < 8; ++r)
      Lg[(size_t)(r + mhi) * SEQ + s0 + n16] = acc[r];
  }
  __syncthreads();

  // ---- Phase 2: exact softmax over each of the 16 rows (fp32, in LDS) ----
  const int row = tid >> 4;
  const int l16 = tid & 15;
  float* Lr = Lg + (size_t)row * SEQ;
  float mx = -3.4e38f;
  for (int s = l16; s < SEQ; s += 16) mx = fmaxf(mx, Lr[s]);
  red[tid] = mx; __syncthreads();
  for (int off = 8; off; off >>= 1) {
    if (l16 < off) red[tid] = fmaxf(red[tid], red[tid + off]);
    __syncthreads();
  }
  mx = red[row << 4];
  __syncthreads();
  float sm = 0.f;
  for (int s = l16; s < SEQ; s += 16) {
    float e = __expf(Lr[s] - mx);
    Lr[s] = e; sm += e;
  }
  red[tid] = sm; __syncthreads();
  for (int off = 8; off; off >>= 1) {
    if (l16 < off) red[tid] += red[tid + off];
    __syncthreads();
  }
  float inv = 1.f / red[row << 4];
  for (int s = l16; s < SEQ; s += 16) Lr[s] *= inv;
  __syncthreads();

  // ---- Phase 3: PV, wave w handles keys [w*512, w*512+512), 4 cc-tiles ----
  v8f pv[4] = {};
  const float* prow = Lg + (size_t)n16 * SEQ;      // A: M=tq row, K=s
  for (int ks = 0; ks < 16; ++ks) {
    int s0 = wave * 512 + ks * 32;
    v16bf pa = load_frag_f32(prow + s0, kHalf);
#pragma unroll
    for (int n = 0; n < 4; ++n) {
      const __bf16* vrow = vB + ((size_t)bh * HDIM + n * 16 + n16) * SEQ + s0;  // B: N=cc, K=s
      pv[n] = wmma_bf16(pa, load_frag(vrow, kHalf), pv[n]);
    }
  }
#pragma unroll
  for (int n = 0; n < 4; ++n)
#pragma unroll
    for (int r = 0; r < 8; ++r)
      red2[(wave * 16 + r + mhi) * 64 + n * 16 + n16] = pv[n][r];
  __syncthreads();

  // cross-wave reduction -> aT [B][T][C] bf16
  for (int i = tid; i < 16 * 64; i += 256) {
    int rr = i >> 6, cc = i & 63;
    float s = 0.f;
#pragma unroll
    for (int w = 0; w < 8; ++w) s += red2[(w * 16 + rr) * 64 + cc];
    int b = bh >> 2, h = bh & 3;
    aT[((size_t)b * SEQ + tq0 + rr) * CHN + h * HDIM + cc] = f2bf(s);
  }
}

// ---------------- Kernel 5: proj GEMM + bias + residual -> out (fp32) -------
// Same 2x2 register-blocked scheme as the QKV GEMM.
__global__ __launch_bounds__(128) void proj_gemm_kernel(
    const __bf16* wproj, const float* proj_b, const __bf16* aT,
    const float* x, float* out) {
  const int b    = blockIdx.z;
  const int lane = threadIdx.x & 31;
  const int wave = threadIdx.x >> 5;
  const int wm   = wave >> 1, wn = wave & 1;
  const int m0   = blockIdx.y * 64 + wm * 32;
  const int t0   = blockIdx.x * 64 + wn * 32;
  const int n16  = lane & 15;
  const int kHalf = (lane & 16) ? 8 : 0;
  const int mhi   = (lane & 16) ? 8 : 0;

  const __bf16* arow0 = wproj + (size_t)(m0 + n16) * CHN;
  const __bf16* arow1 = arow0 + (size_t)16 * CHN;
  const __bf16* brow0 = aT + ((size_t)b * SEQ + t0 + n16) * CHN;
  const __bf16* brow1 = brow0 + (size_t)16 * CHN;

  v8f acc00 = {}, acc01 = {}, acc10 = {}, acc11 = {};
#pragma unroll
  for (int ks = 0; ks < CHN / 32; ++ks) {
    v16bf a0 = load_frag(arow0 + ks * 32, kHalf);
    v16bf a1 = load_frag(arow1 + ks * 32, kHalf);
    v16bf b0 = load_frag(brow0 + ks * 32, kHalf);
    v16bf b1 = load_frag(brow1 + ks * 32, kHalf);
    acc00 = wmma_bf16(a0, b0, acc00);
    acc01 = wmma_bf16(a0, b1, acc01);
    acc10 = wmma_bf16(a1, b0, acc10);
    acc11 = wmma_bf16(a1, b1, acc11);
  }

  v8f accs[2][2] = {{acc00, acc01}, {acc10, acc11}};
#pragma unroll
  for (int i = 0; i < 2; ++i)
#pragma unroll
    for (int j = 0; j < 2; ++j) {
      const int tcol = t0 + j * 16 + n16;
#pragma unroll
      for (int r = 0; r < 8; ++r) {
        int o = m0 + i * 16 + r + mhi;
        size_t idx = ((size_t)b * CHN + o) * SEQ + tcol;
        out[idx] = x[idx] + proj_b[o] + accs[i][j][r];
      }
    }
}

extern "C" void kernel_launch(void* const* d_in, const int* in_sizes, int n_in,
                              void* d_out, int out_size, void* d_ws, size_t ws_size,
                              hipStream_t stream) {
  const float* x      = (const float*)d_in[0];
  const float* gn_w   = (const float*)d_in[1];
  const float* gn_b   = (const float*)d_in[2];
  const float* qkv_w  = (const float*)d_in[3];
  const float* qkv_b  = (const float*)d_in[4];
  const float* proj_w = (const float*)d_in[5];
  const float* proj_b = (const float*)d_in[6];
  float* out = (float*)d_out;

  // Workspace layout (~42.5 MB): stats | wqkv bf16 | wproj bf16 | xnT | qT | kT | v | aT
  char* ws = (char*)d_ws;
  float*  stats = (float*)ws;                              // 256 floats
  __bf16* wqkv  = (__bf16*)(ws + 1024);
  __bf16* wproj = wqkv + 3 * CHN * CHN;
  __bf16* xnT   = wproj + CHN * CHN;
  const size_t PLANE = (size_t)BATCH * SEQ * CHN;          // 4M elems
  __bf16* qT = xnT + PLANE;
  __bf16* kT = qT + PLANE;
  __bf16* vB = kT + PLANE;
  __bf16* aT = vB + PLANE;

  wconv_kernel<<<(3 * CHN * CHN + CHN * CHN + 255) / 256, 256, 0, stream>>>(
      qkv_w, proj_w, wqkv, wproj);
  gn_stats_kernel<<<BATCH * GRP, 256, 0, stream>>>(x, stats);
  gn_apply_kernel<<<(BATCH * CHN * SEQ) / 256, 256, 0, stream>>>(
      x, gn_w, gn_b, stats, xnT);
  qkv_gemm_kernel<<<dim3(SEQ / 64, 3 * CHN / 64, BATCH), 128, 0, stream>>>(
      wqkv, qkv_b, xnT, qT, kT, vB);
  size_t attn_lds = (size_t)(16 * SEQ + 8 * 16 * 64) * sizeof(float);  // 288 KB
  attn_kernel<<<BATCH * HEADS * (SEQ / 16), 256, attn_lds, stream>>>(qT, kT, vB, aT);
  proj_gemm_kernel<<<dim3(SEQ / 64, CHN / 64, BATCH), 128, 0, stream>>>(
      wproj, proj_b, aT, x, out);
}
